// LinkAttModule_39797166965346
// MI455X (gfx1250) — matrix operand order repaired and checked by reading, not compiled, not run on hardware
//
#include <hip/hip_runtime.h>
#include <hip/hip_bf16.h>

typedef __bf16 bf16_t;
typedef __attribute__((ext_vector_type(16))) __bf16 v16bf;
typedef __attribute__((ext_vector_type(8)))  __bf16 v8bf;
typedef __attribute__((ext_vector_type(8)))  float  v8f;
typedef __attribute__((ext_vector_type(4)))  int    i4;

// Address-space qualified views (keep AS explicit so LDS ops lower to ds_*,
// not flat_*, and global->AS casts stay out of the inner loop).
typedef __attribute__((address_space(1))) bf16_t gm_bf16;   // global bf16
typedef __attribute__((address_space(3))) bf16_t ls_bf16;   // LDS bf16
typedef __attribute__((address_space(1))) i4     g_i4;      // global v4i
typedef __attribute__((address_space(3))) i4     l_i4;      // LDS v4i
typedef __attribute__((address_space(3))) v8bf   ls_v8bf;   // LDS 16B bf16 chunk

#define BATCH     4
#define SEQ       2048
#define HIDDEN    1024
#define SCALE_OUT (1.0f / 128.0f)   // mean over 16 heads * 1/sqrt(64)

// GEMM tiling: 128x128 block tile, K-step 32, 8 waves (2x4), 64x32 per wave.
#define BM  128
#define BN  128
#define BK  32
#define LDT 40   // LDS row stride in bf16 elems (32 + 8 pad -> 80B rows, 16B aligned)

// CDNA5 async global->LDS copy path (ASYNCcnt), with sync fallback.
#if __has_builtin(__builtin_amdgcn_global_load_async_to_lds_b128) && \
    __has_builtin(__builtin_amdgcn_s_wait_asynccnt)
#define USE_ASYNC_LDS 1
#else
#define USE_ASYNC_LDS 0
#endif

// ---------------------------------------------------------------------------
// fp32 -> bf16 conversion (vectorized 4-wide)
// ---------------------------------------------------------------------------
__global__ void cvt_f32_to_bf16(const float* __restrict__ src,
                                bf16_t* __restrict__ dst, int n4) {
    int i = blockIdx.x * blockDim.x + threadIdx.x;
    if (i < n4) {
        float4 v = ((const float4*)src)[i];
        bf16_t o0 = (bf16_t)v.x, o1 = (bf16_t)v.y;
        bf16_t o2 = (bf16_t)v.z, o3 = (bf16_t)v.w;
        bf16_t* p = dst + (size_t)i * 4;
        p[0] = o0; p[1] = o1; p[2] = o2; p[3] = o3;
    }
}

// ---------------------------------------------------------------------------
// Shared NT-GEMM core: C[m,n] = sum_k A[m,k] * Bt[n,k]
// A:  [M, K] row-major bf16,  Bt: [N, K] row-major bf16,  K = HIDDEN.
// Block computes a BMxBN tile; each wave owns a 64x32 patch (4x2 WMMA accs).
// Double-buffered LDS fed by GLOBAL_LOAD_ASYNC_TO_LDS_B128 when available.
// ---------------------------------------------------------------------------
union Frag { v16bf v; v8bf h[2]; };

__device__ __forceinline__
void compute_tile(const ls_bf16* AsB, const ls_bf16* BsB,
                  int wr, int wc, int l16, int hk, v8f acc[4][2]) {
    v16bf afr[4], bfr[2];
    #pragma unroll
    for (int mt = 0; mt < 4; ++mt) {
        // A 16x32 (ISA 7.12.2): lane = M (l16); VGPR0-3: K=8*hk+0..7, VGPR4-7: +16
        const ls_bf16* row = AsB + (wr * 64 + mt * 16 + l16) * LDT;
        Frag u;
        u.h[0] = *(const ls_v8bf*)(row + hk * 8);
        u.h[1] = *(const ls_v8bf*)(row + hk * 8 + 16);
        afr[mt] = u.v;
    }
    #pragma unroll
    for (int nt = 0; nt < 2; ++nt) {
        // B 32x16: lane = N (l16); VGPR0-3: K=8*hk+0..7, VGPR4-7: +16
        const ls_bf16* row = BsB + (wc * 32 + nt * 16 + l16) * LDT;
        Frag u;
        u.h[0] = *(const ls_v8bf*)(row + hk * 8);
        u.h[1] = *(const ls_v8bf*)(row + hk * 8 + 16);
        bfr[nt] = u.v;
    }
    #pragma unroll
    for (int mt = 0; mt < 4; ++mt)
        #pragma unroll
        for (int nt = 0; nt < 2; ++nt)
            acc[mt][nt] = __builtin_amdgcn_wmma_f32_16x16x32_bf16(
                false, afr[mt], false, bfr[nt],
                (short)0, acc[mt][nt], false, false);
}

__device__ __forceinline__
void gemm_core(const gm_bf16* __restrict__ A, const gm_bf16* __restrict__ Bt,
               ls_bf16* As, ls_bf16* Bs, int m0, int n0, v8f acc[4][2]) {
    const int tid  = threadIdx.x;
    const int lane = tid & 31;
    const int wave = tid >> 5;
    const int wr   = wave >> 2;      // 0..1 : wave row   (64 M-rows each)
    const int wc   = wave & 3;       // 0..3 : wave col   (32 N-cols each)
    const int l16  = lane & 15;
    const int hk   = lane >> 4;      // half-wave selector (K-group)

    #pragma unroll
    for (int mt = 0; mt < 4; ++mt)
        #pragma unroll
        for (int nt = 0; nt < 2; ++nt)
            acc[mt][nt] = (v8f){0.f,0.f,0.f,0.f,0.f,0.f,0.f,0.f};

    // Tile staging map: 512 chunks of 16B per (A|B) tile, 2 per thread.
    const int q0 = tid, q1 = tid + 256;
    const int ar0 = q0 >> 2, ac0 = (q0 & 3) * 8;
    const int ar1 = q1 >> 2, ac1 = (q1 & 3) * 8;
    const int NK = HIDDEN / BK;

#if USE_ASYNC_LDS
    // Double-buffered: async copy tile kt+1 while WMMA consumes tile kt.
    ls_bf16* Ab[2] = {As, As + BM * LDT};
    ls_bf16* Bb[2] = {Bs, Bs + BN * LDT};

    auto async_tile = [&](ls_bf16* AsB, ls_bf16* BsB, int kb) {
        __builtin_amdgcn_global_load_async_to_lds_b128(
            (g_i4*)(A  + (size_t)(m0 + ar0) * HIDDEN + kb + ac0),
            (l_i4*)(AsB + ar0 * LDT + ac0), 0, 0);
        __builtin_amdgcn_global_load_async_to_lds_b128(
            (g_i4*)(A  + (size_t)(m0 + ar1) * HIDDEN + kb + ac1),
            (l_i4*)(AsB + ar1 * LDT + ac1), 0, 0);
        __builtin_amdgcn_global_load_async_to_lds_b128(
            (g_i4*)(Bt + (size_t)(n0 + ar0) * HIDDEN + kb + ac0),
            (l_i4*)(BsB + ar0 * LDT + ac0), 0, 0);
        __builtin_amdgcn_global_load_async_to_lds_b128(
            (g_i4*)(Bt + (size_t)(n0 + ar1) * HIDDEN + kb + ac1),
            (l_i4*)(BsB + ar1 * LDT + ac1), 0, 0);
    };

    async_tile(Ab[0], Bb[0], 0);
    for (int kt = 0; kt < NK; ++kt) {
        __builtin_amdgcn_s_wait_asynccnt(0);  // this wave's LDS writes landed
        __syncthreads();                      // everyone's tile visible
        if (kt + 1 < NK)
            async_tile(Ab[(kt + 1) & 1], Bb[(kt + 1) & 1], (kt + 1) * BK);
        compute_tile(Ab[kt & 1], Bb[kt & 1], wr, wc, l16, hk, acc);
    }
#else
    // Fallback: global->reg prefetch + single-buffer LDS, two barriers.
    i4 ra0, ra1, rb0, rb1;
    auto gload = [&](int kb) {
        ra0 = *(const g_i4*)(A  + (size_t)(m0 + ar0) * HIDDEN + kb + ac0);
        ra1 = *(const g_i4*)(A  + (size_t)(m0 + ar1) * HIDDEN + kb + ac1);
        rb0 = *(const g_i4*)(Bt + (size_t)(n0 + ar0) * HIDDEN + kb + ac0);
        rb1 = *(const g_i4*)(Bt + (size_t)(n0 + ar1) * HIDDEN + kb + ac1);
    };
    gload(0);
    for (int kt = 0; kt < NK; ++kt) {
        __syncthreads();
        *(l_i4*)(As + ar0 * LDT + ac0) = ra0;
        *(l_i4*)(As + ar1 * LDT + ac1) = ra1;
        *(l_i4*)(Bs + ar0 * LDT + ac0) = rb0;
        *(l_i4*)(Bs + ar1 * LDT + ac1) = rb1;
        if (kt + 1 < NK) gload((kt + 1) * BK);
        __syncthreads();
        compute_tile(As, Bs, wr, wc, l16, hk, acc);
    }
#endif
}

// ---------------------------------------------------------------------------
// Projection: out_bf16[m,n] = bf16( sum_k hs[m,k]*W[n,k] + bias[n] )
// M = BATCH*SEQ = 8192, N = HIDDEN.  Q/K stay RT-cached (re-read from L2).
// ---------------------------------------------------------------------------
__global__ __launch_bounds__(256)
void proj_kernel(const bf16_t* __restrict__ hs, const bf16_t* __restrict__ W,
                 const float* __restrict__ bias, bf16_t* __restrict__ out) {
    __shared__ __align__(16) bf16_t AsRaw[2 * BM * LDT];
    __shared__ __align__(16) bf16_t BsRaw[2 * BN * LDT];
    ls_bf16* As = (ls_bf16*)AsRaw;           // one generic->AS3 cast, hoisted
    ls_bf16* Bs = (ls_bf16*)BsRaw;
    const int m0 = blockIdx.y * BM;
    const int n0 = blockIdx.x * BN;

    v8f acc[4][2];
    gemm_core((const gm_bf16*)hs, (const gm_bf16*)W, As, Bs, m0, n0, acc);

    const int lane = threadIdx.x & 31, wave = threadIdx.x >> 5;
    const int wr = wave >> 2, wc = wave & 3;
    const int l16 = lane & 15, hk = lane >> 4;

    #pragma unroll
    for (int mt = 0; mt < 4; ++mt) {
        const int rbase = m0 + wr * 64 + mt * 16 + hk * 8;  // C: VGPR r -> M = r + 8*hk
        #pragma unroll
        for (int nt = 0; nt < 2; ++nt) {
            const int n = n0 + wc * 32 + nt * 16 + l16;     // C: lane -> N
            const float bv = bias[n];
            #pragma unroll
            for (int r = 0; r < 8; ++r)
                out[(size_t)(rbase + r) * HIDDEN + n] = (bf16_t)(acc[mt][nt][r] + bv);
        }
    }
}

// ---------------------------------------------------------------------------
// Scores: out[b,i,j] = (1/128) * <Q[b,i,:],K[b,j,:]> * mask[b,i] * mask[b,j]
// Output is write-once: nontemporal stores keep Q/K resident in L2.
// ---------------------------------------------------------------------------
__global__ __launch_bounds__(256)
void scores_kernel(const bf16_t* __restrict__ Q, const bf16_t* __restrict__ Kb,
                   const float* __restrict__ mask, float* __restrict__ out) {
    __shared__ __align__(16) bf16_t AsRaw[2 * BM * LDT];
    __shared__ __align__(16) bf16_t BsRaw[2 * BN * LDT];
    ls_bf16* As = (ls_bf16*)AsRaw;
    ls_bf16* Bs = (ls_bf16*)BsRaw;
    const int b  = blockIdx.z;
    const int m0 = blockIdx.y * BM;
    const int n0 = blockIdx.x * BN;
    const gm_bf16* A  = (const gm_bf16*)Q  + (size_t)b * SEQ * HIDDEN;
    const gm_bf16* Bt = (const gm_bf16*)Kb + (size_t)b * SEQ * HIDDEN;

    v8f acc[4][2];
    gemm_core(A, Bt, As, Bs, m0, n0, acc);

    const int lane = threadIdx.x & 31, wave = threadIdx.x >> 5;
    const int wr = wave >> 2, wc = wave & 3;
    const int l16 = lane & 15, hk = lane >> 4;
    const float* mb = mask + (size_t)b * SEQ;
    float* ob = out + (size_t)b * SEQ * SEQ;

    #pragma unroll
    for (int mt = 0; mt < 4; ++mt) {
        const int rbase = m0 + wr * 64 + mt * 16 + hk * 8;
        #pragma unroll
        for (int nt = 0; nt < 2; ++nt) {
            const int j  = n0 + wc * 32 + nt * 16 + l16;
            const float mj = mb[j] * SCALE_OUT;
            #pragma unroll
            for (int r = 0; r < 8; ++r) {
                const int i = rbase + r;
                __builtin_nontemporal_store(acc[mt][nt][r] * mj * mb[i],
                                            &ob[(size_t)i * SEQ + j]);
            }
        }
    }
}

// ---------------------------------------------------------------------------
// Launch
// ---------------------------------------------------------------------------
extern "C" void kernel_launch(void* const* d_in, const int* in_sizes, int n_in,
                              void* d_out, int out_size, void* d_ws, size_t ws_size,
                              hipStream_t stream) {
    const float* hs    = (const float*)d_in[0];   // [B,S,H]
    const float* amask = (const float*)d_in[1];   // [B,S]
    const float* Wq    = (const float*)d_in[2];   // [H,H]
    const float* bq    = (const float*)d_in[3];   // [H]
    const float* Wk    = (const float*)d_in[4];   // [H,H]
    const float* bk    = (const float*)d_in[5];   // [H]
    float* out         = (float*)d_out;           // [B,S,S]

    const size_t nHS = (size_t)BATCH * SEQ * HIDDEN;   // 8.39M elems
    const size_t nW  = (size_t)HIDDEN * HIDDEN;        // 1.05M elems

    bf16_t* hs_b = (bf16_t*)d_ws;
    bf16_t* wq_b = hs_b + nHS;
    bf16_t* wk_b = wq_b + nW;
    bf16_t* q_b  = wk_b + nW;
    bf16_t* k_b  = q_b + nHS;     // total ws use ~54.5 MB (bf16)

    // Phase 0: downconvert operands once (fp32 -> bf16)
    {
        int n4 = (int)(nHS / 4);
        cvt_f32_to_bf16<<<(n4 + 255) / 256, 256, 0, stream>>>(hs, hs_b, n4);
        n4 = (int)(nW / 4);
        cvt_f32_to_bf16<<<(n4 + 255) / 256, 256, 0, stream>>>(Wq, wq_b, n4);
        cvt_f32_to_bf16<<<(n4 + 255) / 256, 256, 0, stream>>>(Wk, wk_b, n4);
    }

    // Phase 1: Q/K projections  (M=8192, N=1024, K=1024)
    dim3 gp(HIDDEN / BN, (BATCH * SEQ) / BM);   // (8, 64)
    proj_kernel<<<gp, 256, 0, stream>>>(hs_b, wq_b, bq, q_b);
    proj_kernel<<<gp, 256, 0, stream>>>(hs_b, wk_b, bk, k_b);

    // Phase 2: batched Q K^T with mean-over-heads collapse + masking
    dim3 gs(SEQ / BN, SEQ / BM, BATCH);         // (16, 16, 4)
    scores_kernel<<<gs, 256, 0, stream>>>(q_b, k_b, amask, out);
}
